// AbstractContrastiveLoss_41377714929832
// MI455X (gfx1250) — matrix operand order repaired and checked by reading, not compiled, not run on hardware
//
#include <hip/hip_runtime.h>

typedef __attribute__((ext_vector_type(16))) _Float16     v16h;
typedef __attribute__((ext_vector_type(8)))  float        v8f;
typedef __attribute__((ext_vector_type(4)))  unsigned int v4u;
typedef __attribute__((ext_vector_type(8)))  int          v8i;
typedef __attribute__((ext_vector_type(4)))  int          v4i;

#define N_  4
#define E_  16
#define H_  768
#define W_  768
#define HW_ (H_ * W_)
#define C_  64
#define DELTA_VAR  0.5f
#define DELTA_DIST 2.0f
#define ALPHA 1.0f
#define BETA  1.0f
#define GAMMA 0.001f
#define EPSF  1e-12f

// workspace layout, in floats
#define WS_SUMS  0                    // N*C*E = 4096
#define WS_CNT   (WS_SUMS + N_*C_*E_) // N*C   = 256
#define WS_MEANS (WS_CNT  + N_*C_)    // N*C*E = 4096
#define WS_VAR   (WS_MEANS + N_*C_*E_)// N
#define WS_DIST  (WS_VAR + N_)        // N
#define WS_REG   (WS_DIST + N_)       // N
#define WS_TOTAL (WS_REG + N_)

#define WAVES_      8
#define SEG_BLOCKS  96    // blocks per sample
#define SEG_ITERS   24    // 96 * 8 * 24 * 32 = 589824 pixels exactly
#define TSTRIDE     36    // LDS tile row stride in dwords (TDM pad 32+4)
#define VAR_BLOCKS  144
#define VAR_QUADS   4     // 144 * 256 * 4 quads * 4 px = 589824 exactly

// ---------------------------------------------------------------- zero
__global__ void k_zero(float* __restrict__ ws) {
  int i = blockIdx.x * blockDim.x + threadIdx.x;
  if (i < WS_TOTAL) ws[i] = 0.0f;
}

// ---------------------------------------------------------------- TDM tile load
// One tensor_load_to_lds per wave: 16 rows x 32 floats, row stride HW_ floats,
// LDS padding 4 dwords after every 32 dwords -> [16][36] layout (16B aligned rows).
__device__ __forceinline__ void tdm_load_tile(const float* gbase, void* lds_ptr) {
  const unsigned lds_off = (unsigned)(size_t)lds_ptr;      // addr[31:0] = LDS offset
  const unsigned long long ga = (unsigned long long)(size_t)gbase;

  v4u g0;
  g0.x = 1u;                                  // count=1 (valid user descriptor)
  g0.y = lds_off;                             // lds_addr
  g0.z = (unsigned)(ga & 0xFFFFFFFFu);        // global_addr[31:0]
  g0.w = (unsigned)((ga >> 32) & 0x01FFFFFFu) // global_addr[56:32]
       | (2u << 30);                          // type = 2 ("image")

  v8i g1;
  g1.s0 = (int)((2u << 16)                    // data_size = 4 bytes
              | (1u << 20)                    // pad_enable
              | (4u << 22)                    // pad_interval = 32 dwords
              | (3u << 25));                  // pad_amount   = 4 dwords
  g1.s1 = (int)(((unsigned)HW_ & 0xFFFFu) << 16);            // tensor_dim0[15:0]
  g1.s2 = (int)((((unsigned)HW_ >> 16) & 0xFFFFu)            // tensor_dim0[31:16]
              | (16u << 16));                                // tensor_dim1 = 16
  g1.s3 = (int)(32u << 16);                   // tensor_dim1 hi = 0, tile_dim0 = 32
  g1.s4 = (int)16u;                           // tile_dim1 = 16, tile_dim2 = 0
  g1.s5 = (int)(unsigned)HW_;                 // tensor_dim0_stride[31:0]
  g1.s6 = 0;                                  // stride0 hi, stride1 lo
  g1.s7 = 0;                                  // stride1 hi

  v4i z4 = {0, 0, 0, 0};
#if defined(__clang_major__) && (__clang_major__ >= 23)
  v8i z8 = {0, 0, 0, 0, 0, 0, 0, 0};
  __builtin_amdgcn_tensor_load_to_lds(g0, g1, z4, z4, z8, 0);
#else
  __builtin_amdgcn_tensor_load_to_lds(g0, g1, z4, z4, 0);
#endif
}

// ---------------------------------------------------------------- segment sum via TDM + WMMA
// sums[C,E] = OneHot(P,C)^T x emb(P,E); counts via LDS atomics.
__global__ __launch_bounds__(256) void k_segsum(const float* __restrict__ inp,
                                                const int*   __restrict__ tgt,
                                                float*       __restrict__ ws) {
  __shared__ __align__(16) float embt[WAVES_][2][E_][TSTRIDE]; // TDM-padded, dbl-buffered
  __shared__ __align__(16) int   tgts[WAVES_][2][32];
  __shared__ unsigned            cnt[C_];
  __shared__ float               bsum[C_ * E_];

  const int n    = blockIdx.x / SEG_BLOCKS;
  const int b    = blockIdx.x % SEG_BLOCKS;
  const int tid  = threadIdx.x;
  const int wave = tid >> 5;
  const int lane = tid & 31;

  for (int i = tid; i < C_ * E_; i += 256) bsum[i] = 0.0f;
  if (tid < C_) cnt[tid] = 0u;
  __syncthreads();

  const int chunk0 = b * (WAVES_ * SEG_ITERS);
  const float* sample = inp + n * E_ * HW_;

  // prologue: issue chunk for it = 0 into buffer 0
  {
    const int p0 = (chunk0 + wave) * 32;
    const int t  = tgt[n * HW_ + p0 + lane];
    tgts[wave][0][lane] = t;
    atomicAdd(&cnt[t], 1u);
    tdm_load_tile(sample + p0, &embt[wave][0][0][0]);
  }

  v8f acc[4] = {};
  const int e_lane = lane & 15;        // B/C/D: lane selects column (N / e)
  const int kbase  = (lane >> 4) * 16; // unused; B lanes 16-31 hold K=8..15, 24..31
  const int khalf  = (lane >> 4) * 8;  // A: 16-bit A 16x32 K mapping
  const int m      = lane & 15;        // A: row (cluster within group)

  for (int it = 0; it < SEG_ITERS; ++it) {
    const int cb = it & 1;
    if (it + 1 < SEG_ITERS) {          // uniform branch: EXEC stays all-1s
      const int p0 = (chunk0 + (it + 1) * WAVES_ + wave) * 32;
      const int t  = tgt[n * HW_ + p0 + lane];
      tgts[wave][cb ^ 1][lane] = t;
      atomicAdd(&cnt[t], 1u);
      tdm_load_tile(sample + p0, &embt[wave][cb ^ 1][0][0]);
      __builtin_amdgcn_s_wait_tensorcnt((short)1);  // current buffer's DMA done
    } else {
      __builtin_amdgcn_s_wait_tensorcnt((short)0);
    }

    // targets this lane needs for its A-fragment positions (2x ds_load_b128 pairs)
    int tk[16];
    {
      const int4* tr = (const int4*)tgts[wave][cb];
      const int   q  = khalf >> 2;          // 0 or 2
      int4 q0 = tr[q + 0];
      int4 q1 = tr[q + 1];
      int4 q2 = tr[q + 4];
      int4 q3 = tr[q + 5];
      tk[0]  = q0.x; tk[1]  = q0.y; tk[2]  = q0.z; tk[3]  = q0.w;
      tk[4]  = q1.x; tk[5]  = q1.y; tk[6]  = q1.z; tk[7]  = q1.w;
      tk[8]  = q2.x; tk[9]  = q2.y; tk[10] = q2.z; tk[11] = q2.w;
      tk[12] = q3.x; tk[13] = q3.y; tk[14] = q3.z; tk[15] = q3.w;
    }

    // B fragment: emb values, lane = e, halves = K (4x ds_load_b128 + cvt_pk)
    v16h bfrag;
#pragma unroll
    for (int h = 0; h < 16; ++h)
      bfrag[h] = (_Float16)embt[wave][cb][e_lane][khalf + (h & 7) + ((h >> 3) << 4)];

    // build all 4 one-hot A fragments first, then issue WMMAs back-to-back
    v16h af[4];
#pragma unroll
    for (int g = 0; g < 4; ++g) {
      const int c = g * 16 + m;
#pragma unroll
      for (int h = 0; h < 16; ++h)
        af[g][h] = (tk[h] == c) ? (_Float16)1.0f : (_Float16)0.0f;
    }
#pragma unroll
    for (int g = 0; g < 4; ++g)
      acc[g] = __builtin_amdgcn_wmma_f32_16x16x32_f16(
          false, af[g], false, bfrag, (short)0, acc[g], false, false);
  }

  // flush C/D fragments (lane 0-15: M=r, lane 16-31: M=r+8; N = lane&15)
  const int mo = (lane >> 4) * 8;
#pragma unroll
  for (int g = 0; g < 4; ++g)
#pragma unroll
    for (int r = 0; r < 8; ++r)
      atomicAdd(&bsum[(g * 16 + mo + r) * E_ + e_lane], acc[g][r]);
  __syncthreads();

  float* sums = ws + WS_SUMS + n * C_ * E_;
  for (int i = tid; i < C_ * E_; i += 256) atomicAdd(&sums[i], bsum[i]);
  if (tid < C_) atomicAdd(ws + WS_CNT + n * C_ + tid, (float)cnt[tid]);
}

// ---------------------------------------------------------------- means + distance + reg
__global__ __launch_bounds__(64) void k_stats(float* __restrict__ ws) {
  __shared__ float ml[C_][E_];
  __shared__ float red0[C_];
  __shared__ float red1[C_];
  const int n = blockIdx.x;
  const int c = threadIdx.x;

  const float* sums  = ws + WS_SUMS  + n * C_ * E_;
  float*       means = ws + WS_MEANS + n * C_ * E_;
  const float  inv   = 1.0f / fmaxf(ws[WS_CNT + n * C_ + c], 1.0f);
#pragma unroll
  for (int e = 0; e < E_; ++e) {
    float mv = sums[c * E_ + e] * inv;
    ml[c][e] = mv;
    means[c * E_ + e] = mv;
  }
  __syncthreads();

  float dsum = 0.0f;
  for (int c2 = 0; c2 < C_; ++c2) {
    if (c2 == c) continue;
    float d2 = EPSF;
#pragma unroll
    for (int e = 0; e < E_; ++e) {
      float df = ml[c][e] - ml[c2][e];
      d2 += df * df;
    }
    float h = fmaxf(2.0f * DELTA_DIST - sqrtf(d2), 0.0f);
    dsum += h * h;
  }
  float r2 = EPSF;
#pragma unroll
  for (int e = 0; e < E_; ++e) r2 += ml[c][e] * ml[c][e];

  red0[c] = dsum;
  red1[c] = sqrtf(r2);
  __syncthreads();
  for (int s = 32; s > 0; s >>= 1) {
    if (c < s) { red0[c] += red0[c + s]; red1[c] += red1[c + s]; }
    __syncthreads();
  }
  if (c == 0) {
    ws[WS_DIST + n] = red0[0];
    ws[WS_REG  + n] = red1[0];
  }
}

// ---------------------------------------------------------------- variance term (streaming, float4)
__global__ __launch_bounds__(256) void k_var(const float* __restrict__ inp,
                                             const int*   __restrict__ tgt,
                                             float*       __restrict__ ws) {
  __shared__ float ml[C_][E_];
  __shared__ float cl[C_];
  __shared__ float red[256];

  const int n   = blockIdx.x / VAR_BLOCKS;
  const int b   = blockIdx.x % VAR_BLOCKS;
  const int tid = threadIdx.x;

  const float* means = ws + WS_MEANS + n * C_ * E_;
  for (int i = tid; i < C_ * E_; i += 256) ml[i >> 4][i & 15] = means[i];
  if (tid < C_) cl[tid] = fmaxf(ws[WS_CNT + n * C_ + tid], 1.0f);
  __syncthreads();

  float local = 0.0f;
  const int stride = VAR_BLOCKS * 256;  // quads per sweep
  for (int i = 0; i < VAR_QUADS; ++i) {
    const int q  = b * 256 + tid + i * stride;
    const int p4 = q * 4;               // 16B-aligned pixel quad
    if (i + 1 < VAR_QUADS)              // global_prefetch_b8 of next tile
      __builtin_prefetch(&inp[n * E_ * HW_ + p4 + stride * 4], 0, 0);

    const int4 tq = *(const int4*)&tgt[n * HW_ + p4];
    float d2x = EPSF, d2y = EPSF, d2z = EPSF, d2w = EPSF;
#pragma unroll
    for (int e = 0; e < E_; ++e) {
      const float4 v = *(const float4*)&inp[(n * E_ + e) * HW_ + p4];
      float dx = v.x - ml[tq.x][e]; d2x += dx * dx;
      float dy = v.y - ml[tq.y][e]; d2y += dy * dy;
      float dz = v.z - ml[tq.z][e]; d2z += dz * dz;
      float dw = v.w - ml[tq.w][e]; d2w += dw * dw;
    }
    float hx = fmaxf(sqrtf(d2x) - DELTA_VAR, 0.0f);
    float hy = fmaxf(sqrtf(d2y) - DELTA_VAR, 0.0f);
    float hz = fmaxf(sqrtf(d2z) - DELTA_VAR, 0.0f);
    float hw = fmaxf(sqrtf(d2w) - DELTA_VAR, 0.0f);
    local += hx * hx / cl[tq.x] + hy * hy / cl[tq.y]
           + hz * hz / cl[tq.z] + hw * hw / cl[tq.w];
  }
  red[tid] = local;
  __syncthreads();
  for (int s = 128; s > 0; s >>= 1) {
    if (tid < s) red[tid] += red[tid + s];
    __syncthreads();
  }
  if (tid == 0) atomicAdd(ws + WS_VAR + n, red[0]);
}

// ---------------------------------------------------------------- finalize
__global__ void k_final(const float* __restrict__ ws, float* __restrict__ out) {
  float acc = 0.0f;
  for (int n = 0; n < N_; ++n) {
    float v = ws[WS_VAR  + n] / (float)C_;
    float d = ws[WS_DIST + n] / (float)(C_ * (C_ - 1));
    float r = ws[WS_REG  + n] / (float)C_;
    acc += ALPHA * v + BETA * d + GAMMA * r;
  }
  out[0] = acc / (float)N_;
}

// ---------------------------------------------------------------- launch
extern "C" void kernel_launch(void* const* d_in, const int* in_sizes, int n_in,
                              void* d_out, int out_size, void* d_ws, size_t ws_size,
                              hipStream_t stream) {
  const float* inp = (const float*)d_in[0];
  const int*   tgt = (const int*)d_in[1];
  float*       ws  = (float*)d_ws;
  float*       out = (float*)d_out;

  hipLaunchKernelGGL(k_zero,   dim3((WS_TOTAL + 255) / 256), dim3(256), 0, stream, ws);
  hipLaunchKernelGGL(k_segsum, dim3(N_ * SEG_BLOCKS),        dim3(256), 0, stream, inp, tgt, ws);
  hipLaunchKernelGGL(k_stats,  dim3(N_),                     dim3(64),  0, stream, ws);
  hipLaunchKernelGGL(k_var,    dim3(N_ * VAR_BLOCKS),        dim3(256), 0, stream, inp, tgt, ws);
  hipLaunchKernelGGL(k_final,  dim3(1),                      dim3(1),   0, stream, ws, out);
}
